// FastGTNs_72902774882367
// MI455X (gfx1250) — compile-verified, matrix operand order checked by reading, not run on hardware
//
#include <hip/hip_runtime.h>
#include <hip/hip_bf16.h>

typedef __attribute__((ext_vector_type(16))) _Float16 v16h;
typedef __attribute__((ext_vector_type(8)))  float    v8f;

namespace {
constexpr int   kT   = 4;    // graph types
constexpr int   kC   = 2;    // channels
constexpr int   kH   = 64;   // hidden width
constexpr int   kFin = 128;  // input features
constexpr int   kL   = 2;    // GT layers per FastGTN
constexpr float kBeta = 0.1f;
}

// ---------------------------------------------------------------------------
// filt[l,c,:] = softmax(conv_w[l,c,:]) over T=4. 16 values total.
// ---------------------------------------------------------------------------
__global__ void softmax_filt_kernel(const float* __restrict__ cw,
                                    float* __restrict__ filt, int rows) {
  int i = blockIdx.x * blockDim.x + threadIdx.x;
  if (i >= rows) return;
  float v[kT];
  float m = -1e30f;
  #pragma unroll
  for (int t = 0; t < kT; ++t) { v[t] = cw[i * kT + t]; m = fmaxf(m, v[t]); }
  float sum = 0.f;
  #pragma unroll
  for (int t = 0; t < kT; ++t) { v[t] = __expf(v[t] - m); sum += v[t]; }
  float inv = 1.f / sum;
  #pragma unroll
  for (int t = 0; t < kT; ++t) filt[i * kT + t] = v[t] * inv;
}

// ---------------------------------------------------------------------------
// Batched GEMM on WMMA: Cout[c] = A (fp32 [M,K]) @ B[c] (fp32 [K,ncols]).
// One wave per 16x16 output tile; f16 inputs, f32 accumulate.
// A tile layout (16-bit A 16x32): lane m=lane&15, half=lane>>4;
//   a[j]   <- K = kc + 8*half + j        (j=0..7)
//   a[8+j] <- K = kc + 16 + 8*half + j
// B tile mirrors with lane&15 = column. C/D: acc[r] = C[r + 8*half][lane&15].
// ---------------------------------------------------------------------------
__global__ void gemm_f32_wmma_kernel(const float* __restrict__ A,
                                     const float* __restrict__ B,
                                     float* __restrict__ Cout,
                                     int M, int K, int ncols,
                                     long long bStrideB, long long bStrideC) {
  int c = blockIdx.y;
  B    += (long long)c * bStrideB;
  Cout += (long long)c * bStrideC;

  int wid   = threadIdx.x >> 5;
  int lane  = threadIdx.x & 31;
  int tile  = blockIdx.x * (blockDim.x >> 5) + wid;
  int tilesN = ncols >> 4;
  int tm = tile / tilesN, tn = tile % tilesN;
  if (tm * 16 >= M) return;

  int mr = lane & 15, half = lane >> 4;
  const float* arow = A + (long long)(tm * 16 + mr) * K;
  v8f acc = {};
  for (int kc = 0; kc < K; kc += 32) {
    v16h a, b;
    const float* ap = arow + kc + 8 * half;
    #pragma unroll
    for (int j = 0; j < 8; ++j) {
      a[j]     = (_Float16)ap[j];
      a[8 + j] = (_Float16)ap[16 + j];
    }
    const float* bp = B + (long long)(kc + 8 * half) * ncols + tn * 16 + mr;
    #pragma unroll
    for (int j = 0; j < 8; ++j) {
      b[j]     = (_Float16)bp[(long long)j * ncols];
      b[8 + j] = (_Float16)bp[(long long)(16 + j) * ncols];
    }
    acc = __builtin_amdgcn_wmma_f32_16x16x32_f16(false, a, false, b,
                                                 (short)0, acc, false, false);
  }
  float* crow = Cout + (long long)(tm * 16 + 8 * half) * ncols + tn * 16 + mr;
  #pragma unroll
  for (int r = 0; r < 8; ++r) crow[(long long)r * ncols] = acc[r];
}

// ---------------------------------------------------------------------------
// out = relu(A (f16 [M,K]) @ B (fp32 [K,ncols]) + bias). Same WMMA tiling.
// ---------------------------------------------------------------------------
__global__ void gemm_lin_relu_kernel(const _Float16* __restrict__ A,
                                     const float* __restrict__ B,
                                     const float* __restrict__ bias,
                                     float* __restrict__ Cout,
                                     int M, int K, int ncols) {
  int wid   = threadIdx.x >> 5;
  int lane  = threadIdx.x & 31;
  int tile  = blockIdx.x * (blockDim.x >> 5) + wid;
  int tilesN = ncols >> 4;
  int tm = tile / tilesN, tn = tile % tilesN;
  if (tm * 16 >= M) return;

  int mr = lane & 15, half = lane >> 4;
  const _Float16* arow = A + (long long)(tm * 16 + mr) * K;
  float bv = bias[tn * 16 + mr];
  v8f acc = {};
  for (int kc = 0; kc < K; kc += 32) {
    v16h a, b;
    const _Float16* ap = arow + kc + 8 * half;
    #pragma unroll
    for (int j = 0; j < 8; ++j) {
      a[j]     = ap[j];
      a[8 + j] = ap[16 + j];
    }
    const float* bp = B + (long long)(kc + 8 * half) * ncols + tn * 16 + mr;
    #pragma unroll
    for (int j = 0; j < 8; ++j) {
      b[j]     = (_Float16)bp[(long long)j * ncols];
      b[8 + j] = (_Float16)bp[(long long)(16 + j) * ncols];
    }
    acc = __builtin_amdgcn_wmma_f32_16x16x32_f16(false, a, false, b,
                                                 (short)0, acc, false, false);
  }
  float* crow = Cout + (long long)(tm * 16 + 8 * half) * ncols + tn * 16 + mr;
  #pragma unroll
  for (int r = 0; r < 8; ++r) {
    float v = acc[r] + bv;
    crow[(long long)r * ncols] = v > 0.f ? v : 0.f;
  }
}

// ---------------------------------------------------------------------------
// SpMM over the raw edge list (dense_A never materialized; adj@H == sum over
// edges of filt[c,t]*w_e * H[d,:] scattered into row s). One wave per
// (channel, graph-type, edge); lane i owns 2 of the 64 features.
// H buffers are 1 MB/channel -> L2 resident; atomics land in L2.
// ---------------------------------------------------------------------------
__global__ void spmm_edges_kernel(const int* __restrict__ ei,   // [T,2,E]
                                  const float* __restrict__ ew, // [T,E]
                                  const float* __restrict__ filt, // [C,T]
                                  const float* __restrict__ Hin,  // [C,N,kH]
                                  float* __restrict__ Hout,       // [C,N,kH]
                                  int E, int N) {
  long long gid = (long long)blockIdx.x * blockDim.x + threadIdx.x;
  long long gw  = gid >> 5;
  int lane = threadIdx.x & 31;
  long long TE = (long long)kT * E;
  if (gw >= (long long)kC * TE) return;
  int c = (int)(gw / TE);
  long long rem = gw - (long long)c * TE;
  int t = (int)(rem / E);
  int e = (int)(rem - (long long)t * E);

  int s = ei[((long long)t * 2    ) * E + e];
  int d = ei[((long long)t * 2 + 1) * E + e];
  float w = ew[(long long)t * E + e] * filt[c * kT + t];

  const float* src = Hin  + ((long long)c * N + d) * kH + lane * 2;
  float*       dst = Hout + ((long long)c * N + s) * kH + lane * 2;
  float x = w * src[0];
  float y = w * src[1];
  __hip_atomic_fetch_add(dst,     x, __ATOMIC_RELAXED, __HIP_MEMORY_SCOPE_AGENT);
  __hip_atomic_fetch_add(dst + 1, y, __ATOMIC_RELAXED, __HIP_MEMORY_SCOPE_AGENT);
}

// ---------------------------------------------------------------------------
// Hcat[n, c*kH+h] = (f16) relu(beta*H0[c,n,h] + (1-beta)*H2[c,n,h])
// (fused residual-mix + ReLU + [C,N,H] -> [N, C*H] transpose, f16 output so
//  the following linear layer feeds WMMA directly.)
// ---------------------------------------------------------------------------
__global__ void mix_relu_pack_kernel(const float* __restrict__ H0,
                                     const float* __restrict__ H2,
                                     _Float16* __restrict__ Hcat, int N) {
  long long i = (long long)blockIdx.x * blockDim.x + threadIdx.x;
  long long total = (long long)N * kC * kH;
  if (i >= total) return;
  int n    = (int)(i / (kC * kH));
  int rest = (int)(i - (long long)n * (kC * kH));
  int c = rest / kH, h = rest - c * kH;
  long long src = ((long long)c * N + n) * kH + h;
  float v = kBeta * H0[src] + (1.0f - kBeta) * H2[src];
  v = v > 0.f ? v : 0.f;
  Hcat[i] = (_Float16)v;
}

// ---------------------------------------------------------------------------
extern "C" void kernel_launch(void* const* d_in, const int* in_sizes, int n_in,
                              void* d_out, int out_size, void* d_ws, size_t ws_size,
                              hipStream_t stream) {
  const int*   ei  = (const int*)d_in[0];
  const float* ew  = (const float*)d_in[1];
  const float* X   = (const float*)d_in[2];
  const float* Ws1 = (const float*)d_in[3];
  const float* cw1 = (const float*)d_in[4];
  const float* lw1 = (const float*)d_in[5];
  const float* lb1 = (const float*)d_in[6];
  const float* Ws2 = (const float*)d_in[7];
  const float* cw2 = (const float*)d_in[8];
  const float* lw2 = (const float*)d_in[9];
  const float* lb2 = (const float*)d_in[10];

  const int N = in_sizes[2] / kFin;  // 4096
  const int E = in_sizes[1] / kT;    // 131072

  // workspace carve-up (floats)
  const size_t chn = (size_t)kC * N * kH;   // per-buffer [C,N,kH]
  float*     ws   = (float*)d_ws;
  float*     filt = ws;                      // kL*kC*kT = 16
  float*     H0   = ws + 64;                 // X@Ws (also the detached X_)
  float*     HA   = H0 + chn;                // layer-0 output
  float*     HB   = HA + chn;                // layer-1 output
  _Float16*  Hcat = (_Float16*)(HB + chn);   // [N, kC*kH] f16
  float*     Xmid = (float*)(Hcat + (size_t)N * kC * kH);  // FastGTN#1 out

  const int gemmTiles   = (N / 16) * (kH / 16);           // 1024 waves
  const int gemmBlocks  = gemmTiles / 4;                  // 4 waves/block
  const long long edgeWaves = (long long)kC * kT * E;     // 1,048,576
  const int spmmBlocks  = (int)((edgeWaves * 32) / 256);  // 131,072
  const int packTotal   = N * kC * kH;

  auto run_fast_gtn = [&](const float* Xin, int K, const float* Ws,
                          const float* cw, const float* lw, const float* lb,
                          float* outp) {
    softmax_filt_kernel<<<1, 32, 0, stream>>>(cw, filt, kL * kC);

    // H0[c] = Xin @ Ws[c]   (WMMA)
    gemm_f32_wmma_kernel<<<dim3(gemmBlocks, kC), 128, 0, stream>>>(
        Xin, Ws, H0, N, K, kH, (long long)K * kH, (long long)N * kH);

    // conv layer 0: HA = adj(filt[0]) @ H0  via edge scatter
    hipMemsetAsync(HA, 0, chn * sizeof(float), stream);
    spmm_edges_kernel<<<spmmBlocks, 256, 0, stream>>>(ei, ew, filt, H0, HA, E, N);

    // conv layer 1: HB = adj(filt[1]) @ HA
    hipMemsetAsync(HB, 0, chn * sizeof(float), stream);
    spmm_edges_kernel<<<spmmBlocks, 256, 0, stream>>>(ei, ew, filt + kC * kT,
                                                      HA, HB, E, N);

    // Hcat = relu(beta*H0 + (1-beta)*HB), transposed+packed to f16
    mix_relu_pack_kernel<<<packTotal / 256, 256, 0, stream>>>(H0, HB, Hcat, N);

    // outp = relu(Hcat @ lw + lb)   (WMMA)
    gemm_lin_relu_kernel<<<gemmBlocks, 128, 0, stream>>>(Hcat, lw, lb, outp,
                                                         N, kC * kH, kH);
  };

  run_fast_gtn(X,    kFin, Ws1, cw1, lw1, lb1, Xmid);
  run_fast_gtn(Xmid, kH,   Ws2, cw2, lw2, lb2, (float*)d_out);
}
